// PatchAttentionLayer_9887014716269
// MI455X (gfx1250) — compile-verified, matrix operand order checked by reading, not run on hardware
//
#include <hip/hip_runtime.h>

typedef __bf16 bf16_t;
typedef __attribute__((ext_vector_type(8)))  __bf16 v8bf;
typedef __attribute__((ext_vector_type(16))) __bf16 v16bf;
typedef __attribute__((ext_vector_type(8)))  float  v8f;

#define CC   256
#define NN   4096      // h*w
#define BB   8
#define MTOT 32768.0f  // BB*NN samples per channel for BN stats
#define EPSV 1e-5f
#define ATTN_SCALE 0.125f  // 64^-0.5

// ---------------------------------------------------------------------------
// Fragment helpers (CDNA5 WMMA bf16 16x16x32, wave32)
// A-layout (ISA 7.12.2): lane L -> row = L&15; K base = (L<16)?0:8;
//   VGPR0..3 = K base+0..7, VGPR4..7 = K base+16..23.
// Loads two 16B chunks per lane from a row-major [16 x >=32] bf16 tile.
// ---------------------------------------------------------------------------
__device__ __forceinline__ v16bf load_frag(const bf16_t* base, int ld, int lane) {
    int row = lane & 15;
    int kb  = (lane & 16) ? 8 : 0;
    const bf16_t* p = base + row * ld + kb;
    v8bf lo = *(const v8bf*)(p);        // K = kb .. kb+7
    v8bf hi = *(const v8bf*)(p + 16);   // K = kb+16 .. kb+23
    return __builtin_shufflevector(lo, hi, 0,1,2,3,4,5,6,7,8,9,10,11,12,13,14,15);
}

__device__ __forceinline__ v8f wmma_bf16(v16bf a, v16bf b, v8f c) {
    return __builtin_amdgcn_wmma_f32_16x16x32_bf16(
        /*neg_a=*/false, a, /*neg_b=*/false, b,
        /*c_mod=*/(short)0, c, /*reuse_a=*/false, /*reuse_b=*/false);
}

// ---------------------------------------------------------------------------
// Kernel 0: zero BN stat accumulators + convert weights fp32 -> bf16
// grid = 768 x 256  (3*256*256 weight elements)
// ---------------------------------------------------------------------------
__global__ void prep_kernel(const float* __restrict__ Wq,
                            const float* __restrict__ Wk,
                            const float* __restrict__ Wv,
                            float* __restrict__ stats,      // 1536 floats (sum|sumsq)
                            bf16_t* __restrict__ Wbf) {     // 3*65536 bf16
    int idx = blockIdx.x * blockDim.x + threadIdx.x;
    if (idx < 1536) stats[idx] = 0.0f;
    int p = idx >> 16;
    int e = idx & 65535;
    const float* W = (p == 0) ? Wq : ((p == 1) ? Wk : Wv);
    Wbf[idx] = (bf16_t)W[e];
}

// ---------------------------------------------------------------------------
// Kernel 1: 1x1-conv projection GEMM (WMMA) + per-channel BN stat reduction
// grid = (64 m-tiles, 8 batch, 3 proj) x 256 threads (8 waves)
// Y[o, m-tile] = W[o,:] @ x[:, m-tile] + bias  -> bf16 workspace
// ---------------------------------------------------------------------------
__global__ void __launch_bounds__(256)
proj_kernel(const float* __restrict__ x,
            const float* __restrict__ bq, const float* __restrict__ bk,
            const float* __restrict__ bv,
            const bf16_t* __restrict__ Wbf,
            bf16_t* __restrict__ Ybf,
            float* __restrict__ statsSum, float* __restrict__ statsSq) {
    __shared__ __attribute__((aligned(16))) bf16_t lds_x[64 * 264]; // x^T tile [m][c]
    __shared__ float lds_sum[CC];
    __shared__ float lds_sq[CC];

    const int tid  = threadIdx.x;
    const int lane = tid & 31;
    const int wave = tid >> 5;
    const int qn   = lane & 15;
    const int hh   = lane >> 4;      // half-wave
    const int m0   = blockIdx.x * 64;
    const int bb   = blockIdx.y;
    const int proj = blockIdx.z;

    const float*  bias = (proj == 0) ? bq : ((proj == 1) ? bk : bv);
    const bf16_t* W    = Wbf + proj * (CC * CC);
    const float*  xb   = x + (size_t)bb * (CC * NN);

    lds_sum[tid] = 0.0f;
    lds_sq[tid]  = 0.0f;

    // stage x tile transposed: thread = channel c, 64 m values, fp32 -> bf16
    {
        const int c = tid;
        const float* xr = xb + (size_t)c * NN + m0;
        #pragma unroll
        for (int m = 0; m < 64; m += 4) {
            float4 v = *(const float4*)(xr + m);
            bf16_t* d = &lds_x[m * 264 + c];
            d[0]     = (bf16_t)v.x;
            d[264]   = (bf16_t)v.y;
            d[528]   = (bf16_t)v.z;
            d[792]   = (bf16_t)v.w;
        }
    }
    __syncthreads();

    // wave owns output channels [wave*32, wave*32+32), all 4 m-subtiles
    v8f acc[2][4] = {};
    #pragma unroll
    for (int cs = 0; cs < 8; ++cs) {                 // K-dim: c in steps of 32
        v16bf a0 = load_frag(W + (wave * 32 +  0) * CC + cs * 32, CC, lane);
        v16bf a1 = load_frag(W + (wave * 32 + 16) * CC + cs * 32, CC, lane);
        #pragma unroll
        for (int q = 0; q < 4; ++q) {
            v16bf bfr = load_frag(lds_x + q * 16 * 264 + cs * 32, 264, lane);
            acc[0][q] = wmma_bf16(a0, bfr, acc[0][q]);
            acc[1][q] = wmma_bf16(a1, bfr, acc[1][q]);
        }
    }

    // epilogue: bias, store bf16 Y, reduce BN stats
    const size_t ybase = ((size_t)(proj * BB + bb) * CC) * NN;
    #pragma unroll
    for (int cs = 0; cs < 2; ++cs) {
        #pragma unroll
        for (int i = 0; i < 8; ++i) {
            const int o = wave * 32 + cs * 16 + i + (hh << 3);
            const float bo = bias[o];
            float s1 = 0.0f, s2 = 0.0f;
            #pragma unroll
            for (int q = 0; q < 4; ++q) {
                float y = acc[cs][q][i] + bo;
                Ybf[ybase + (size_t)o * NN + m0 + q * 16 + qn] = (bf16_t)y;
                s1 += y;
                s2 += y * y;
            }
            #pragma unroll
            for (int msk = 1; msk < 16; msk <<= 1) {   // reduce within half-wave
                s1 += __shfl_xor(s1, msk, 32);
                s2 += __shfl_xor(s2, msk, 32);
            }
            if (qn == 0) {
                atomicAdd(&lds_sum[o], s1);
                atomicAdd(&lds_sq[o],  s2);
            }
        }
    }
    __syncthreads();
    atomicAdd(&statsSum[proj * CC + tid], lds_sum[tid]);
    atomicAdd(&statsSq [proj * CC + tid], lds_sq[tid]);
}

// ---------------------------------------------------------------------------
// Kernel 2: fold BN into per-channel affine (a, b).  grid = 3 x 256
// ---------------------------------------------------------------------------
__global__ void finalize_kernel(const float* __restrict__ statsSum,
                                const float* __restrict__ statsSq,
                                const float* __restrict__ gq, const float* __restrict__ betaq,
                                const float* __restrict__ gk, const float* __restrict__ betak,
                                const float* __restrict__ gv, const float* __restrict__ betav,
                                float* __restrict__ normA, float* __restrict__ normB) {
    int idx = blockIdx.x * blockDim.x + threadIdx.x;   // 0..767
    int p = idx >> 8, c = idx & 255;
    const float* g  = (p == 0) ? gq    : ((p == 1) ? gk    : gv);
    const float* be = (p == 0) ? betaq : ((p == 1) ? betak : betav);
    float mu  = statsSum[idx] * (1.0f / MTOT);
    float var = statsSq[idx] * (1.0f / MTOT) - mu * mu;
    float a   = g[c] * rsqrtf(var + EPSV);
    normA[idx] = a;
    normB[idx] = be[c] - mu * a;
}

// ---------------------------------------------------------------------------
// Kernel 3: apply BN affine in place on bf16 Y.  8 elements/thread.
// grid = 12288 x 256
// ---------------------------------------------------------------------------
__global__ void norm_kernel(bf16_t* __restrict__ Ybf,
                            const float* __restrict__ normA,
                            const float* __restrict__ normB) {
    int t   = blockIdx.x * blockDim.x + threadIdx.x;   // chunk of 8 bf16
    int row = t >> 9;                                  // (t*8)/4096
    int p   = row >> 11;                               // / (8*256)
    int c   = row & 255;
    float a = normA[p * CC + c];
    float b = normB[p * CC + c];
    v8bf v = *(v8bf*)(Ybf + (size_t)t * 8);
    #pragma unroll
    for (int i = 0; i < 8; ++i) v[i] = (bf16_t)((float)v[i] * a + b);
    *(v8bf*)(Ybf + (size_t)t * 8) = v;
}

// ---------------------------------------------------------------------------
// Kernel 4: fused attention  out[:,q] = scale * V @ (K^T @ Q[:,q])
// grid = (64 q-tiles, 8 batch) x 256 threads (8 waves)
// k-loop in blocks of 32, S kept block-local (LDS), out acc in VGPRs.
// ---------------------------------------------------------------------------
__global__ void __launch_bounds__(256)
attn_kernel(const bf16_t* __restrict__ Ybf, float* __restrict__ out) {
    __shared__ __attribute__((aligned(16))) bf16_t lds_q [64 * 264]; // Q^T [q][c]
    __shared__ __attribute__((aligned(16))) bf16_t lds_k [32 * 264]; // K^T [k][c]
    __shared__ __attribute__((aligned(16))) bf16_t lds_st[64 * 40];  // S^T [q][k]

    const int tid  = threadIdx.x;
    const int lane = tid & 31;
    const int wave = tid >> 5;
    const int qn   = lane & 15;
    const int hh   = lane >> 4;
    const int q0   = blockIdx.x * 64;
    const int bb   = blockIdx.y;

    const bf16_t* Qb = Ybf + ((size_t)(0 * BB + bb) * CC) * NN;
    const bf16_t* Kb = Ybf + ((size_t)(1 * BB + bb) * CC) * NN;
    const bf16_t* Vb = Ybf + ((size_t)(2 * BB + bb) * CC) * NN;

    // stage Q^T once: thread = channel c
    {
        const bf16_t* src = Qb + (size_t)tid * NN + q0;
        #pragma unroll
        for (int q8 = 0; q8 < 64; q8 += 8) {
            v8bf v = *(const v8bf*)(src + q8);
            #pragma unroll
            for (int j = 0; j < 8; ++j) lds_q[(q8 + j) * 264 + tid] = v[j];
        }
    }

    const int ks = wave & 1;    // S-phase tile: k subtile
    const int qs = wave >> 1;   // S-phase tile: q subtile
    v8f acc[2][4] = {};         // out accumulators: [c-sub within wave][q-sub]

    for (int kb = 0; kb < NN; kb += 32) {
        // stage K^T block: thread = channel c
        {
            const bf16_t* srck = Kb + (size_t)tid * NN + kb;
            #pragma unroll
            for (int k8 = 0; k8 < 32; k8 += 8) {
                v8bf v = *(const v8bf*)(srck + k8);
                #pragma unroll
                for (int j = 0; j < 8; ++j) lds_k[(k8 + j) * 264 + tid] = v[j];
            }
            if (kb + 32 < NN)
                __builtin_prefetch(Kb + (size_t)tid * NN + kb + 32, 0, 1);
        }
        __syncthreads();

        // S-phase: S[k,q] = sum_c K^T[k,c] * Q[c,q]   (one 16x16 tile per wave)
        v8f s = {0.f, 0.f, 0.f, 0.f, 0.f, 0.f, 0.f, 0.f};
        #pragma unroll
        for (int cs = 0; cs < 8; ++cs) {
            v16bf a = load_frag(lds_k + ks * 16 * 264 + cs * 32, 264, lane);
            v16bf b = load_frag(lds_q + qs * 16 * 264 + cs * 32, 264, lane);
            s = wmma_bf16(a, b, s);
        }
        // store scaled S^T as bf16: D layout -> k = ks*16 + i + hh*8, q = qs*16 + qn
        #pragma unroll
        for (int i = 0; i < 8; ++i) {
            int kk = ks * 16 + i + (hh << 3);
            lds_st[(qs * 16 + qn) * 40 + kk] = (bf16_t)(s[i] * ATTN_SCALE);
        }
        __syncthreads();

        // out-phase: wave owns channels [wave*32, wave*32+32)
        v16bf av0 = load_frag(Vb + (size_t)(wave * 32 +  0) * NN + kb, NN, lane);
        v16bf av1 = load_frag(Vb + (size_t)(wave * 32 + 16) * NN + kb, NN, lane);
        #pragma unroll
        for (int q = 0; q < 4; ++q) {
            v16bf bs = load_frag(lds_st + q * 16 * 40, 40, lane);
            acc[0][q] = wmma_bf16(av0, bs, acc[0][q]);
            acc[1][q] = wmma_bf16(av1, bs, acc[1][q]);
        }
    }

    // epilogue: write fp32 output [b][c][q]
    #pragma unroll
    for (int cs = 0; cs < 2; ++cs) {
        #pragma unroll
        for (int i = 0; i < 8; ++i) {
            const int c = wave * 32 + cs * 16 + i + (hh << 3);
            #pragma unroll
            for (int q = 0; q < 4; ++q) {
                out[((size_t)(bb * CC + c)) * NN + q0 + q * 16 + qn] = acc[cs][q][i];
            }
        }
    }
}

// ---------------------------------------------------------------------------
// Launcher
// ---------------------------------------------------------------------------
extern "C" void kernel_launch(void* const* d_in, const int* in_sizes, int n_in,
                              void* d_out, int out_size, void* d_ws, size_t ws_size,
                              hipStream_t stream) {
    const float* x     = (const float*)d_in[0];
    const float* Wq    = (const float*)d_in[1];
    const float* bq    = (const float*)d_in[2];
    const float* gq    = (const float*)d_in[3];
    const float* betaq = (const float*)d_in[4];
    const float* Wk    = (const float*)d_in[5];
    const float* bk    = (const float*)d_in[6];
    const float* gk    = (const float*)d_in[7];
    const float* betak = (const float*)d_in[8];
    const float* Wv    = (const float*)d_in[9];
    const float* bv    = (const float*)d_in[10];
    const float* gv    = (const float*)d_in[11];
    const float* betav = (const float*)d_in[12];
    float* out = (float*)d_out;

    char* ws = (char*)d_ws;
    float*  statsSum = (float*)(ws);                 //  768 f32
    float*  statsSq  = (float*)(ws + 3072);          //  768 f32
    float*  normA    = (float*)(ws + 6144);          //  768 f32
    float*  normB    = (float*)(ws + 9216);          //  768 f32
    bf16_t* Wbf      = (bf16_t*)(ws + 12288);        //  3*65536 bf16
    bf16_t* Ybf      = (bf16_t*)(ws + 12288 + 393216); // 3*8*256*4096 bf16 (~48 MB)

    prep_kernel<<<768, 256, 0, stream>>>(Wq, Wk, Wv, statsSum, Wbf);

    proj_kernel<<<dim3(64, 8, 3), 256, 0, stream>>>(x, bq, bk, bv, Wbf, Ybf,
                                                    statsSum, statsSq);

    finalize_kernel<<<3, 256, 0, stream>>>(statsSum, statsSq,
                                           gq, betaq, gk, betak, gv, betav,
                                           normA, normB);

    norm_kernel<<<12288, 256, 0, stream>>>(Ybf, normA, normB);

    attn_kernel<<<dim3(64, 8), 256, 0, stream>>>(Ybf, out);
}
